// PhaseBindingLanguageModel_84825604096595
// MI455X (gfx1250) — matrix operand order repaired and checked by reading, not compiled, not run on hardware
//
#include <hip/hip_runtime.h>
#include <math.h>

// ---------------- constants ----------------
#define BB 2
#define LL 2048
#define DD 256
#define KK 32
#define NLAYER 4
#define VOC 32000
#define MM (BB*LL)   // 4096 rows

// ---------------- helpers ----------------
__device__ __forceinline__ unsigned short f2bf(float f) {
    unsigned int u = __float_as_uint(f);
    unsigned int r = (u + 0x7FFFu + ((u >> 16) & 1u)) >> 16;   // RNE
    return (unsigned short)r;
}

// 16B per-lane async copy global -> LDS (tracked on ASYNCcnt, no VGPR round trip).
// Low 32 bits of a generic pointer to LDS are the LDS byte offset (aperture mapping).
__device__ __forceinline__ void async_cp16(void* lds_dst, const void* gsrc) {
    unsigned int       l = (unsigned int)(uintptr_t)lds_dst;
    unsigned long long g = (unsigned long long)(uintptr_t)gsrc;
    asm volatile("global_load_async_to_lds_b128 %0, %1, off" :: "v"(l), "v"(g) : "memory");
}
__device__ __forceinline__ void wait_async0() {
    asm volatile("s_wait_asynccnt 0" ::: "memory");
}

// ---------------- elementwise kernels ----------------
__global__ void cvt_bf16_k(const float* __restrict__ in, unsigned short* __restrict__ out, int n) {
    int i = blockIdx.x * blockDim.x + threadIdx.x;
    if (i < n) out[i] = f2bf(in[i]);
}

// in: [Din, Dout] f32  ->  out: [Dout, Din] bf16   (B^T layout for WMMA loads)
__global__ void tcvt_k(const float* __restrict__ in, unsigned short* __restrict__ out,
                       int Din, int Dout, int n) {
    int j = blockIdx.x * blockDim.x + threadIdx.x;
    if (j >= n) return;
    int o   = j / Din;
    int iin = j - o * Din;
    out[j] = f2bf(in[(size_t)iin * Dout + o]);
}

__global__ void embed_k(const int* __restrict__ x, const float* __restrict__ emb,
                        const float* __restrict__ pos, float* __restrict__ h, int total) {
    int i = blockIdx.x * blockDim.x + threadIdx.x;
    if (i >= total) return;
    int d  = i & (DD - 1);
    int bl = i / DD;
    int l  = bl & (LL - 1);
    int tok = x[bl];
    h[i] = emb[(size_t)tok * DD + d] + pos[(size_t)l * DD + d];
}

// phase = tanh(p)*pi + l*10000^(-k/K);  amp = softplus(a)+0.1
__global__ void phasor_k(const float* __restrict__ p, const float* __restrict__ a,
                         float* __restrict__ re, float* __restrict__ im, int total) {
    int i = blockIdx.x * blockDim.x + threadIdx.x;
    if (i >= total) return;
    int k = i & (KK - 1);
    int l = (i / KK) & (LL - 1);
    float freq  = __expf(-(float)k * (9.210340371976184f / (float)KK));  // 10000^(-k/K)
    float phase = tanhf(p[i]) * 3.14159265358979323846f + (float)l * freq;
    float aa  = a[i];
    float amp = fmaxf(aa, 0.f) + log1pf(__expf(-fabsf(aa))) + 0.1f;
    re[i] = amp * cosf(phase);
    im[i] = amp * sinf(phase);
}

// ---------------- layernorm (row = D=256, block = 256 threads) ----------------
__global__ __launch_bounds__(256)
void layernorm_k(const float* __restrict__ in, const float* __restrict__ g,
                 const float* __restrict__ b, float* __restrict__ out) {
    __shared__ float rs[8], rq[8], stat[2];
    int row = blockIdx.x;
    int tid = threadIdx.x;
    float x = in[(size_t)row * DD + tid];
    float s = x, q = x * x;
    #pragma unroll
    for (int o = 16; o > 0; o >>= 1) { s += __shfl_xor(s, o); q += __shfl_xor(q, o); }
    if ((tid & 31) == 0) { rs[tid >> 5] = s; rq[tid >> 5] = q; }
    __syncthreads();
    if (tid == 0) {
        float S = 0.f, Q = 0.f;
        #pragma unroll
        for (int i = 0; i < 8; ++i) { S += rs[i]; Q += rq[i]; }
        float mu  = S * (1.f / DD);
        float var = Q * (1.f / DD) - mu * mu;
        stat[0] = mu; stat[1] = rsqrtf(var + 1e-5f);
    }
    __syncthreads();
    out[(size_t)row * DD + tid] = (x - stat[0]) * stat[1] * g[tid] + b[tid];
}

// ---------------- WMMA bf16 GEMM:  C[M,N] = A[M,Kd] * Bt[N,Kd]^T (+bias)(tanh)(+res) ----------------
// block = 256 threads = 8 waves (4 along M x 2 along N); wave tile 16(M) x 64(N).
// Double-buffered k-loop: frags of step i+1 load into a distinct register set while
// step i's 4 WMMAs issue -> partial s_wait_loadcnt instead of full-drain waits.
__global__ __launch_bounds__(256)
void gemm_bf16_k(const unsigned short* __restrict__ Ap, const unsigned short* __restrict__ Btp,
                 const float* __restrict__ bias, const float* __restrict__ res,
                 float* __restrict__ C, int M, int N, int Kd, int act) {
    typedef __bf16 bf16x16 __attribute__((ext_vector_type(16)));
    typedef float  f32x8   __attribute__((ext_vector_type(8)));
    union Frag { bf16x16 v; uint4 q[2]; };

    const __bf16* A  = (const __bf16*)Ap;
    const __bf16* Bt = (const __bf16*)Btp;

    const int tid  = threadIdx.x;
    const int lane = tid & 31;
    const int w    = tid >> 5;
    const int wm   = w & 3;          // 4 waves along M
    const int wn   = w >> 2;         // 2 waves along N
    const int m0   = blockIdx.x * 64 + wm * 16;
    const int n0   = blockIdx.y * 128 + wn * 64;

    const int lo = lane & 15, hi = lane >> 4;
    int arow = m0 + lo; if (arow > M - 1) arow = M - 1;
    const __bf16* ap = A + (size_t)arow * Kd;

    const __bf16* bp[4];
    #pragma unroll
    for (int t = 0; t < 4; ++t) {
        int brow = n0 + t * 16 + lo; if (brow > N - 1) brow = N - 1;
        bp[t] = Bt + (size_t)brow * Kd;
    }

    // A 16x32 bf16 frag, contiguous k order chosen here (matches B): lanes 0-15 hold K=[0..15]; lanes 16-31 hold K=[16..31]
    // B 32x16 bf16 frag, contiguous k order chosen here (matches A): lanes 0-15 hold K=[0..15]; lanes 16-31 hold K=[16..31] (col = lane&15)
    auto load_frags = [&](int kk, Frag& a, Frag* b) {
        a.q[0] = *(const uint4*)(ap + kk + hi * 16);
        a.q[1] = *(const uint4*)(ap + kk + hi * 16 + 8);
        #pragma unroll
        for (int t = 0; t < 4; ++t) {
            b[t].q[0] = *(const uint4*)(bp[t] + kk + hi * 16);
            b[t].q[1] = *(const uint4*)(bp[t] + kk + hi * 16 + 8);
        }
    };

    f32x8 acc[4] = {};
    Frag a0, b0[4], a1, b1[4];
    load_frags(0, a0, b0);
    for (int kk = 0; kk < Kd; kk += 64) {
        if (kk + 32 < Kd) load_frags(kk + 32, a1, b1);        // prefetch odd step
        #pragma unroll
        for (int t = 0; t < 4; ++t)
            acc[t] = __builtin_amdgcn_wmma_f32_16x16x32_bf16(false, a0.v, false, b0[t].v,
                                                             (short)0, acc[t], false, false);
        if (kk + 64 < Kd) load_frags(kk + 64, a0, b0);        // prefetch next even step
        if (kk + 32 < Kd) {
            #pragma unroll
            for (int t = 0; t < 4; ++t)
                acc[t] = __builtin_amdgcn_wmma_f32_16x16x32_bf16(false, a1.v, false, b1[t].v,
                                                                 (short)0, acc[t], false, false);
        }
    }

    // C layout: lane col = lane&15 within tile; VGPR j -> row m0 + j + 8*(lane>>4)
    #pragma unroll
    for (int t = 0; t < 4; ++t) {
        int col = n0 + t * 16 + lo;
        if (col >= N) continue;
        #pragma unroll
        for (int j = 0; j < 8; ++j) {
            int row = m0 + j + hi * 8;
            if (row >= M) continue;
            float xv = acc[t][j];
            if (bias) xv += bias[col];
            if (act)  xv = tanhf(xv);
            if (res)  xv += res[(size_t)row * N + col];
            C[(size_t)row * N + col] = xv;
        }
    }
}

// ---------------- causal linear-attention scan ----------------
// retrieved[b,l,d] = sum_{t<=l} sum_k (kre[t,k]qre[l,k] + kim[t,k]qim[l,k]) v[t,d] / sqrt((l+1)K)
// grid = B * (D/64); thread (kg,dd): owns k in [8kg,8kg+8), d = d0+dd; state in registers.
// 16-step chunks of k/q/v staged via global_load_async_to_lds_b128 (ASYNCcnt).
__global__ __launch_bounds__(256)
void scan_k(const float* __restrict__ kre, const float* __restrict__ kim,
            const float* __restrict__ qre, const float* __restrict__ qim,
            const float* __restrict__ v,   float* __restrict__ retr) {
    __shared__ float skre[16 * KK], skim[16 * KK], sqre[16 * KK], sqim[16 * KK], sv[16 * 64];
    const int nseg = DD / 64;
    const int b  = blockIdx.x / nseg;
    const int d0 = (blockIdx.x % nseg) * 64;
    const int tid = threadIdx.x;
    const int kg = tid & 3, dd = tid >> 2;

    float sre[8] = {}, sim[8] = {};
    for (int c0 = 0; c0 < LL; c0 += 16) {
        size_t kbase = ((size_t)b * LL + c0) * KK;   // float index into k/q arrays
        {   // 768 16B chunks total = 3 per thread, statically assigned (no divergent ptr arrays)
            int idx = (tid & 127) * 4;               // float offset within 512-float region
            if (tid < 128) async_cp16(skre + idx, kre + kbase + idx);
            else           async_cp16(skim + idx, kim + kbase + idx);
            if (tid < 128) async_cp16(sqre + idx, qre + kbase + idx);
            else           async_cp16(sqim + idx, qim + kbase + idx);
            int l = tid >> 4, c16 = (tid & 15) * 4;  // 16 rows x 16 chunks of the V slice
            async_cp16(sv + l * 64 + c16,
                       v + ((size_t)b * LL + c0 + l) * DD + d0 + c16);
        }
        wait_async0();
        __syncthreads();
        for (int t = 0; t < 16; ++t) {
            float vv = sv[t * 64 + dd];
            float r = 0.f;
            #pragma unroll
            for (int j = 0; j < 8; ++j) {
                int k = kg * 8 + j;
                sre[j] = fmaf(skre[t * KK + k], vv, sre[j]);
                sim[j] = fmaf(skim[t * KK + k], vv, sim[j]);
                r = fmaf(sre[j], sqre[t * KK + k], r);
                r = fmaf(sim[j], sqim[t * KK + k], r);
            }
            r += __shfl_xor(r, 1);
            r += __shfl_xor(r, 2);
            if (kg == 0)
                retr[((size_t)b * LL + c0 + t) * DD + d0 + dd] =
                    r * rsqrtf((float)(c0 + t + 1) * (float)KK);
        }
        __syncthreads();
    }
}

// ---------------- host orchestration ----------------
extern "C" void kernel_launch(void* const* d_in, const int* in_sizes, int n_in,
                              void* d_out, int out_size, void* d_ws, size_t ws_size,
                              hipStream_t stream) {
    (void)in_sizes; (void)n_in; (void)out_size; (void)ws_size;
    const int*   x     = (const int*)  d_in[0];
    const float* embed = (const float*)d_in[1];
    const float* pos   = (const float*)d_in[2];
    const float* headb = (const float*)d_in[3];
    const float* ng    = (const float*)d_in[4];
    const float* nb    = (const float*)d_in[5];
    const float* ln_g  = (const float*)d_in[6];
    const float* ln_b  = (const float*)d_in[7];
    const float* kW1   = (const float*)d_in[8];
    const float* kb1   = (const float*)d_in[9];
    const float* kW2   = (const float*)d_in[10];
    const float* kb2   = (const float*)d_in[11];
    const float* kWa   = (const float*)d_in[12];
    const float* kba   = (const float*)d_in[13];
    const float* qW1   = (const float*)d_in[14];
    const float* qb1   = (const float*)d_in[15];
    const float* qW2   = (const float*)d_in[16];
    const float* qb2   = (const float*)d_in[17];
    const float* qWa   = (const float*)d_in[18];
    const float* qba   = (const float*)d_in[19];
    const float* vW    = (const float*)d_in[20];
    const float* vb    = (const float*)d_in[21];
    const float* olng  = (const float*)d_in[22];
    const float* olnb  = (const float*)d_in[23];
    const float* oW    = (const float*)d_in[24];
    const float* ob    = (const float*)d_in[25];

    char* wsb = (char*)d_ws;
    size_t off = 0;
    auto take = [&](size_t bytes) -> void* {
        void* p = wsb + off;
        off = (off + bytes + 255) & ~(size_t)255;
        return p;
    };
    float* h    = (float*)take(sizeof(float) * MM * DD);
    float* z    = (float*)take(sizeof(float) * MM * DD);
    float* t1   = (float*)take(sizeof(float) * MM * DD);
    float* vbuf = (float*)take(sizeof(float) * MM * DD);
    float* retr = (float*)take(sizeof(float) * MM * DD);
    float* rln  = (float*)take(sizeof(float) * MM * DD);
    float* pk   = (float*)take(sizeof(float) * MM * KK);
    float* pa   = (float*)take(sizeof(float) * MM * KK);
    float* kre  = (float*)take(sizeof(float) * MM * KK);
    float* kim  = (float*)take(sizeof(float) * MM * KK);
    float* qre  = (float*)take(sizeof(float) * MM * KK);
    float* qim  = (float*)take(sizeof(float) * MM * KK);
    unsigned short* zb   = (unsigned short*)take(2ull * MM * DD);
    unsigned short* t1b  = (unsigned short*)take(2ull * MM * DD);
    unsigned short* rlnb = (unsigned short*)take(2ull * MM * DD);
    unsigned short* embb = (unsigned short*)take(2ull * VOC * DD);
    const size_t WLSZ = 4ull * DD * DD + 4ull * DD * KK;   // bf16 elems per layer
    unsigned short* wbuf = (unsigned short*)take(2ull * NLAYER * WLSZ);

    auto gemm = [&](const unsigned short* A, const unsigned short* Bt, const float* bias,
                    const float* resid, float* C, int n, int kd, int act) {
        dim3 grid(MM / 64, (n + 127) / 128);
        gemm_bf16_k<<<grid, 256, 0, stream>>>(A, Bt, bias, resid, C, MM, n, kd, act);
    };
    auto cvt = [&](const float* in, unsigned short* out, int n) {
        cvt_bf16_k<<<(n + 255) / 256, 256, 0, stream>>>(in, out, n);
    };

    // ---- prep: bf16 embedding table + transposed bf16 weights ----
    cvt(embed, embb, VOC * DD);
    for (int i = 0; i < NLAYER; ++i) {
        unsigned short* wl = wbuf + (size_t)i * WLSZ;
        tcvt_k<<<(DD * DD + 255) / 256, 256, 0, stream>>>(kW1 + (size_t)i * DD * DD, wl + 0 * DD * DD, DD, DD, DD * DD);
        tcvt_k<<<(DD * DD + 255) / 256, 256, 0, stream>>>(qW1 + (size_t)i * DD * DD, wl + 1 * DD * DD, DD, DD, DD * DD);
        tcvt_k<<<(DD * DD + 255) / 256, 256, 0, stream>>>(vW  + (size_t)i * DD * DD, wl + 2 * DD * DD, DD, DD, DD * DD);
        tcvt_k<<<(DD * DD + 255) / 256, 256, 0, stream>>>(oW  + (size_t)i * DD * DD, wl + 3 * DD * DD, DD, DD, DD * DD);
        unsigned short* ws2 = wl + 4 * DD * DD;
        tcvt_k<<<(DD * KK + 255) / 256, 256, 0, stream>>>(kW2 + (size_t)i * DD * KK, ws2 + 0 * DD * KK, DD, KK, DD * KK);
        tcvt_k<<<(DD * KK + 255) / 256, 256, 0, stream>>>(kWa + (size_t)i * DD * KK, ws2 + 1 * DD * KK, DD, KK, DD * KK);
        tcvt_k<<<(DD * KK + 255) / 256, 256, 0, stream>>>(qW2 + (size_t)i * DD * KK, ws2 + 2 * DD * KK, DD, KK, DD * KK);
        tcvt_k<<<(DD * KK + 255) / 256, 256, 0, stream>>>(qWa + (size_t)i * DD * KK, ws2 + 3 * DD * KK, DD, KK, DD * KK);
    }

    // ---- embedding gather ----
    embed_k<<<(MM * DD + 255) / 256, 256, 0, stream>>>(x, embed, pos, h, MM * DD);

    // ---- layers ----
    for (int i = 0; i < NLAYER; ++i) {
        const unsigned short* wl  = wbuf + (size_t)i * WLSZ;
        const unsigned short* w1k = wl + 0 * DD * DD;
        const unsigned short* w1q = wl + 1 * DD * DD;
        const unsigned short* wv  = wl + 2 * DD * DD;
        const unsigned short* wo  = wl + 3 * DD * DD;
        const unsigned short* w2k = wl + 4 * DD * DD + 0 * DD * KK;
        const unsigned short* wak = wl + 4 * DD * DD + 1 * DD * KK;
        const unsigned short* w2q = wl + 4 * DD * DD + 2 * DD * KK;
        const unsigned short* waq = wl + 4 * DD * DD + 3 * DD * KK;

        layernorm_k<<<MM, 256, 0, stream>>>(h, ln_g + (size_t)i * DD, ln_b + (size_t)i * DD, z);
        cvt(z, zb, MM * DD);

        // k phasor path
        gemm(zb, w1k, kb1 + (size_t)i * DD, nullptr, t1, DD, DD, 1);          // tanh(z@W1+b1)
        cvt(t1, t1b, MM * DD);
        gemm(t1b, w2k, kb2 + (size_t)i * KK, nullptr, pk, KK, DD, 0);
        gemm(zb, wak, kba + (size_t)i * KK, nullptr, pa, KK, DD, 0);
        phasor_k<<<(MM * KK + 255) / 256, 256, 0, stream>>>(pk, pa, kre, kim, MM * KK);

        // q phasor path
        gemm(zb, w1q, qb1 + (size_t)i * DD, nullptr, t1, DD, DD, 1);
        cvt(t1, t1b, MM * DD);
        gemm(t1b, w2q, qb2 + (size_t)i * KK, nullptr, pk, KK, DD, 0);
        gemm(zb, waq, qba + (size_t)i * KK, nullptr, pa, KK, DD, 0);
        phasor_k<<<(MM * KK + 255) / 256, 256, 0, stream>>>(pk, pa, qre, qim, MM * KK);

        // values + causal scan
        gemm(zb, wv, vb + (size_t)i * DD, nullptr, vbuf, DD, DD, 0);
        scan_k<<<BB * (DD / 64), 256, 0, stream>>>(kre, kim, qre, qim, vbuf, retr);

        // output projection with residual: h = z + LN(retr)@oW + ob
        layernorm_k<<<MM, 256, 0, stream>>>(retr, olng + (size_t)i * DD, olnb + (size_t)i * DD, rln);
        cvt(rln, rlnb, MM * DD);
        gemm(rlnb, wo, ob + (size_t)i * DD, z, h, DD, DD, 0);
    }

    // ---- final norm + tied head ----
    layernorm_k<<<MM, 256, 0, stream>>>(h, ng, nb, z);
    cvt(z, zb, MM * DD);
    gemm(zb, embb, headb, nullptr, (float*)d_out, VOC, DD, 0);
}
